// RGBuvHistBlock_16149077033127
// MI455X (gfx1250) — matrix operand order repaired and checked by reading, not compiled, not run on hardware
//
#include <hip/hip_runtime.h>

// RGB-uv histogram (RGBuvHistBlock) for MI455X / gfx1250.
//
// hist[l,c,u,v] = sum_n du[l,n,c,u] * dv[l,n,c,v]  ==  per (l,c):
//   C(64x64) = Du^T (64xN) x Dv (Nx64), K = 65536  -> f16 WMMA, f32 accum.
// Phase 1 (kernel-weight generation, ~100M rcp-based evals) is the bound;
// each thread computes 2 pixels so all LDS stores are packed b32 and global
// loads are b64. Phase 2 batches all fragment ds_load_b128s of a k-step
// before the 4 back-to-back v_wmma_f32_16x16x32_f16.

typedef __attribute__((ext_vector_type(16))) _Float16 v16h;
typedef __attribute__((ext_vector_type(8)))  float    v8f;

#define H       64
#define NPIX    65536
#define LCOUNT  12              // 4 images x 3 channels
#define KCHUNK  256             // pixels staged in LDS per iteration (2/thread)
#define NSPLIT  32              // workgroups per (l,c)
#define NCHUNK  (NPIX / NSPLIT) // 2048 pixels per workgroup
#define NITER   (NCHUNK / KCHUNK) // 8
#define SIG2    (0.02f * 0.02f)
#define EPSF    1e-6f
#define LN2F    0.69314718056f
#define OUTN    (LCOUNT * H * H) // 49152

union F16Frag { v16h v; uint4 q[2]; };
union H2Pack  { _Float16 h[2]; unsigned int w; };

__device__ __forceinline__ float iqk(float d) {       // sigma^2/(sigma^2+d^2)
  return SIG2 * __builtin_amdgcn_rcpf(fmaf(d, d, SIG2));
}

__global__ void __launch_bounds__(256)
zero_kernel(float* __restrict__ hist) {
  int i = blockIdx.x * blockDim.x + threadIdx.x;
  if (i < OUTN) hist[i] = 0.0f;
}

__global__ void __launch_bounds__(128)
hist_wmma_kernel(const float* __restrict__ x, float* __restrict__ hist) {
  // du/dv tiles: 64 bins x 256 k-values, f16, [bin][k] (k contiguous) so
  // WMMA fragments are aligned ds_load_b128s. 64 KB total (<320 KB/WGP).
  __shared__ __align__(16) _Float16 s_du[H * KCHUNK];
  __shared__ __align__(16) _Float16 s_dv[H * KCHUNK];

  const int tid  = threadIdx.x;
  const int lane = tid & 31;
  const int wave = tid >> 5;       // u-tile row owned by this wave (0..3)
  const int m    = lane & 15;
  const int hi   = lane >> 4;      // lane half (K-group select per ISA layout)

  const int lc    = blockIdx.x / NSPLIT;     // (l,c) pair
  const int split = blockIdx.x - lc * NSPLIT;
  const int l     = lc / 3;
  const int c     = lc - l * 3;
  const float* __restrict__ xl = x + (size_t)l * 3 * NPIX;
  const int n0 = split * NCHUNK;

  v8f acc[4] = {};   // 4 v-tiles (16x16 f32 each) for this wave's u-tile row

  for (int it = 0; it < NITER; ++it) {
    // ---- Phase 1: two pixels/thread -> 128 packed-f16 b32 LDS stores ----
    const int n = n0 + it * KCHUNK + 2 * tid;
    const float2 rp = *(const float2*)&xl[0 * NPIX + n];
    const float2 gp = *(const float2*)&xl[1 * NPIX + n];
    const float2 bp = *(const float2*)&xl[2 * NPIX + n];

    float r0 = fminf(fmaxf(rp.x * 0.5f + 0.5f, 0.0f), 1.0f);
    float g0 = fminf(fmaxf(gp.x * 0.5f + 0.5f, 0.0f), 1.0f);
    float b0 = fminf(fmaxf(bp.x * 0.5f + 0.5f, 0.0f), 1.0f);
    float r1 = fminf(fmaxf(rp.y * 0.5f + 0.5f, 0.0f), 1.0f);
    float g1 = fminf(fmaxf(gp.y * 0.5f + 0.5f, 0.0f), 1.0f);
    float b1 = fminf(fmaxf(bp.y * 0.5f + 0.5f, 0.0f), 1.0f);

    float lr0 = __builtin_amdgcn_logf(r0 + EPSF);  // log2; ln via *LN2 on diff
    float lg0 = __builtin_amdgcn_logf(g0 + EPSF);
    float lb0 = __builtin_amdgcn_logf(b0 + EPSF);
    float lr1 = __builtin_amdgcn_logf(r1 + EPSF);
    float lg1 = __builtin_amdgcn_logf(g1 + EPSF);
    float lb1 = __builtin_amdgcn_logf(b1 + EPSF);

    float iu0, iv0, iu1, iv1;
    if (c == 0) {
      iu0 = (lr0 - lg0) * LN2F; iv0 = (lr0 - lb0) * LN2F;
      iu1 = (lr1 - lg1) * LN2F; iv1 = (lr1 - lb1) * LN2F;
    } else if (c == 1) {
      iu0 = (lg0 - lr0) * LN2F; iv0 = (lg0 - lb0) * LN2F;
      iu1 = (lg1 - lr1) * LN2F; iv1 = (lg1 - lb1) * LN2F;
    } else {
      iu0 = (lb0 - lr0) * LN2F; iv0 = (lb0 - lg0) * LN2F;
      iu1 = (lb1 - lr1) * LN2F; iv1 = (lb1 - lg1) * LN2F;
    }
    float iy0 = __builtin_amdgcn_sqrtf(fmaf(r0, r0, fmaf(g0, g0, fmaf(b0, b0, EPSF))));
    float iy1 = __builtin_amdgcn_sqrtf(fmaf(r1, r1, fmaf(g1, g1, fmaf(b1, b1, EPSF))));

#pragma unroll 8
    for (int u = 0; u < H; ++u) {
      float lin = -3.0f + (float)u * (6.0f / 63.0f);
      H2Pack pu, pv;
      pu.h[0] = (_Float16)(iy0 * iqk(iu0 - lin));
      pu.h[1] = (_Float16)(iy1 * iqk(iu1 - lin));
      pv.h[0] = (_Float16)iqk(iv0 - lin);
      pv.h[1] = (_Float16)iqk(iv1 - lin);
      *(unsigned int*)&s_du[u * KCHUNK + 2 * tid] = pu.w;
      *(unsigned int*)&s_dv[u * KCHUNK + 2 * tid] = pv.w;
    }
    __syncthreads();

    // ---- Phase 2: 8 K-steps; batch 10 ds_load_b128 then 4 WMMAs ----
    // A (16x32 f16): lane<16 -> K {0..15}, lane>=16 -> K {16..31}.
    // B (32x16 f16): lane<16 -> K {0..15}, lane>=16 -> K {16..31}.
    const _Float16* arow = &s_du[(wave * 16 + m) * KCHUNK];
    const _Float16* brow = &s_dv[m * KCHUNK];
#pragma unroll
    for (int s = 0; s < 8; ++s) {
      const int ka = s * 32 + hi * 16;
      const int kb = s * 32 + hi * 16;
      F16Frag a, f0, f1, f2, f3;
      a.q[0]  = *(const uint4*)(arow + ka);
      a.q[1]  = *(const uint4*)(arow + ka + 8);
      f0.q[0] = *(const uint4*)(brow + kb);
      f0.q[1] = *(const uint4*)(brow + kb + 8);
      f1.q[0] = *(const uint4*)(brow + 16 * KCHUNK + kb);
      f1.q[1] = *(const uint4*)(brow + 16 * KCHUNK + kb + 8);
      f2.q[0] = *(const uint4*)(brow + 32 * KCHUNK + kb);
      f2.q[1] = *(const uint4*)(brow + 32 * KCHUNK + kb + 8);
      f3.q[0] = *(const uint4*)(brow + 48 * KCHUNK + kb);
      f3.q[1] = *(const uint4*)(brow + 48 * KCHUNK + kb + 8);
      acc[0] = __builtin_amdgcn_wmma_f32_16x16x32_f16(
          false, a.v, false, f0.v, (short)0, acc[0], false, false);
      acc[1] = __builtin_amdgcn_wmma_f32_16x16x32_f16(
          false, a.v, false, f1.v, (short)0, acc[1], false, false);
      acc[2] = __builtin_amdgcn_wmma_f32_16x16x32_f16(
          false, a.v, false, f2.v, (short)0, acc[2], false, false);
      acc[3] = __builtin_amdgcn_wmma_f32_16x16x32_f16(
          false, a.v, false, f3.v, (short)0, acc[3], false, false);
    }
    __syncthreads();
  }

  // ---- Combine partial 64x64 histograms across the 32 N-splits ----
  // C/D layout: VGPR j, lane -> row = j + 8*hi, col = m (per ISA 7.12.2).
  float* __restrict__ hbase = hist + lc * (H * H);
#pragma unroll
  for (int vt = 0; vt < 4; ++vt) {
#pragma unroll
    for (int j = 0; j < 8; ++j) {
      int row = wave * 16 + hi * 8 + j;
      int col = vt * 16 + m;
      __hip_atomic_fetch_add(&hbase[row * H + col], acc[vt][j],
                             __ATOMIC_RELAXED, __HIP_MEMORY_SCOPE_AGENT);
    }
  }
}

__global__ void __launch_bounds__(256)
norm_kernel(const float* __restrict__ hist, float* __restrict__ out) {
  __shared__ float red[256];
  const int tid = threadIdx.x;
  float s = 0.0f;
  for (int i = tid; i < OUTN; i += 256) s += hist[i];
  red[tid] = s;
  __syncthreads();
#pragma unroll
  for (int off = 128; off > 0; off >>= 1) {
    if (tid < off) red[tid] += red[tid + off];
    __syncthreads();
  }
  const float inv = 1.0f / (red[0] + EPSF);
  for (int i = tid; i < OUTN; i += 256) out[i] = hist[i] * inv;
}

extern "C" void kernel_launch(void* const* d_in, const int* in_sizes, int n_in,
                              void* d_out, int out_size, void* d_ws, size_t ws_size,
                              hipStream_t stream) {
  const float* x  = (const float*)d_in[0];
  float* out      = (float*)d_out;
  float* hist     = (float*)d_ws;   // 49152 floats of scratch

  zero_kernel<<<(OUTN + 255) / 256, 256, 0, stream>>>(hist);
  hist_wmma_kernel<<<LCOUNT * NSPLIT, 128, 0, stream>>>(x, hist);
  norm_kernel<<<1, 256, 0, stream>>>(hist, out);
}
